// CBModel_19344532701217
// MI455X (gfx1250) — matrix operand order repaired and checked by reading, not compiled, not run on hardware
//
#include <hip/hip_runtime.h>

// ============================================================================
// CDNA5 / gfx1250 wave32 bf16-WMMA implementation of the relation-aware
// graph-transformer forward pass.  Round 2: + TDM (tensor_load_to_lds) for the
// edge-bias tables, + double-buffered GEMM mainloop.
//
// Input pointer order assumption (JAX pytree flatten: dict keys sorted):
//   [0]=e_features [1]=edge_bias_key [2]=edge_bias_value
//   [3..66] layers[l] x16 sorted: b1,b2,bk,bo,bq,bv,ln1_b,ln1_g,ln2_b,ln2_g,
//                                 w1,w2,wk,wo,wq,wv
//   [67]=lin_b [68]=lin_w [69]=linear0_b [70]=linear0_w
//   [71]=r_features [72]=type_encoding
//   [73]=target_triple [74]=h_neighbors [75]=t_neighbors [76]=n [77]=adj
// ============================================================================

typedef __bf16 bf16;
typedef __attribute__((ext_vector_type(16))) __bf16 bf16x16;
typedef __attribute__((ext_vector_type(8)))  __bf16 bf16x8;
typedef __attribute__((ext_vector_type(8)))  float  f32x8;
typedef __attribute__((ext_vector_type(4)))  unsigned u32x4;
typedef __attribute__((ext_vector_type(8)))  unsigned u32x8;

#define BB     16
#define LTOK   65
#define LPAD   80     // 5 tiles of 16
#define EFFD   1024
#define NHEADS 16
#define HDIM   64
#define DD     256
#define MM     32
#define SCALE_C 0.125f  // 1/sqrt(64)

// ---------------------------------------------------------------------------
// WMMA helpers (wave32). A: 16x32 bf16 (M x K), B: 32x16 bf16 (K x N, loaded
// from an N-major [n][k] LDS tile), C/D: 16x16 f32.
// ---------------------------------------------------------------------------
__device__ __forceinline__ f32x8 wmma_bf16(bf16x16 a, bf16x16 b, f32x8 c) {
  return __builtin_amdgcn_wmma_f32_16x16x32_bf16(false, a, false, b,
                                                 (short)0, c, false, false);
}

__device__ __forceinline__ bf16x16 frag_a(const bf16* tile, int m0, int ld,
                                          int k0, int lane) {
  const bf16* p = tile + (size_t)(m0 + (lane & 15)) * ld + k0 + ((lane >> 4) << 3);
  bf16x8 lo = *(const bf16x8*)(p);        // k = half*8 + 0..7
  bf16x8 hi = *(const bf16x8*)(p + 16);   // k = 16 + half*8 + 0..7
  bf16x16 f;
#pragma unroll
  for (int i = 0; i < 8; ++i) { f[i] = lo[i]; f[i + 8] = hi[i]; }
  return f;
}

__device__ __forceinline__ bf16x16 frag_b(const bf16* tileT, int n0, int ld,
                                          int k0, int lane) {
  const bf16* p = tileT + (size_t)(n0 + (lane & 15)) * ld + k0 + ((lane >> 4) << 3);
  bf16x8 lo = *(const bf16x8*)(p);        // k = half*8 + 0..7
  bf16x8 hi = *(const bf16x8*)(p + 16);   // k = 16 + half*8 + 0..7
  bf16x16 f;
#pragma unroll
  for (int i = 0; i < 8; ++i) { f[i] = lo[i]; f[i + 8] = hi[i]; }
  return f;
}

// ---------------------------------------------------------------------------
// Tensor Data Mover: 2D f32 tile (width x rows, row stride in elements) from
// global memory into LDS (rows packed contiguously).  D# per ISA 8.3/8.4.
// Issued once per calling wave; caller gates to one wave and then does
// s_wait_tensorcnt(0) + block barrier.
// ---------------------------------------------------------------------------
__device__ __forceinline__ void tdm_load_2d_f32(unsigned lds_off,
                                                const float* gptr,
                                                unsigned width_elems,
                                                unsigned height_rows,
                                                unsigned row_stride_elems) {
  unsigned long long ga = (unsigned long long)gptr;
  u32x4 g0;
  g0[0] = 1u;                                   // count=1 (valid), user mode
  g0[1] = lds_off;                              // lds_addr (bytes)
  g0[2] = (unsigned)(ga & 0xffffffffu);         // global_addr[31:0]
  g0[3] = (unsigned)((ga >> 32) & 0x1ffffffu)   // global_addr[56:32]
          | (2u << 30);                         // type = 2 ("image")
  u32x8 g1;
  g1[0] = (2u << 16);                           // data_size=2 (4 bytes)
  g1[1] = (width_elems & 0xffffu) << 16;        // tensor_dim0[15:0]
  g1[2] = (width_elems >> 16) |                 // tensor_dim0[31:16]
          ((height_rows & 0xffffu) << 16);      // tensor_dim1[15:0]
  g1[3] = (height_rows >> 16) |                 // tensor_dim1[31:16]
          ((width_elems & 0xffffu) << 16);      // tile_dim0
  g1[4] = (height_rows & 0xffffu);              // tile_dim1 (tile_dim2=0 -> 2D)
  g1[5] = row_stride_elems;                     // tensor_dim0_stride[31:0]
  g1[6] = 0;                                    // stride0 hi / stride1 lo
  g1[7] = 0;
  asm volatile("tensor_load_to_lds %0, %1" ::"s"(g0), "s"(g1) : "memory");
}

// ---------------------------------------------------------------------------
// Generic GEMM: C[M,N] = act( A[M,K](f32,row-major,lda) @ B[K,N](f32,row-major)
//                             + bias[N] (+ R[M,N] if RESID) )
// 128 threads = 4 waves; 64x64 tile; BK=32; double-buffered LDS mainloop.
// ---------------------------------------------------------------------------
struct TileRegs {
  float4 a[4];
  float  b[4][4];
};

__device__ __forceinline__ void gemm_load_tile(TileRegs& t, const float* A,
                                               int lda, const float* B, int N,
                                               int row0, int col0, int k0,
                                               int M, int tid) {
#pragma unroll
  for (int i = 0; i < 4; ++i) {
    int chunk = tid + i * 128;
    int r = chunk >> 3;
    int c = (chunk & 7) << 2;
    float4 va = {0.f, 0.f, 0.f, 0.f};
    if (row0 + r < M) va = *(const float4*)(A + (size_t)(row0 + r) * lda + k0 + c);
    t.a[i] = va;
  }
#pragma unroll
  for (int i = 0; i < 4; ++i) {
    int chunk = tid + i * 128;
    int kk = chunk >> 4;
    int n4 = (chunk & 15) << 2;
    const float* src = B + (size_t)(k0 + kk) * N + col0 + n4;
#pragma unroll
    for (int j = 0; j < 4; ++j)
      t.b[i][j] = (col0 + n4 + j < N) ? src[j] : 0.f;
  }
}

__device__ __forceinline__ void gemm_store_tile(const TileRegs& t, bf16* As,
                                                bf16* Bs, int tid) {
#pragma unroll
  for (int i = 0; i < 4; ++i) {
    int chunk = tid + i * 128;
    int r = chunk >> 3;
    int c = (chunk & 7) << 2;
    bf16* dst = As + r * 32 + c;
    dst[0] = (bf16)t.a[i].x; dst[1] = (bf16)t.a[i].y;
    dst[2] = (bf16)t.a[i].z; dst[3] = (bf16)t.a[i].w;
  }
#pragma unroll
  for (int i = 0; i < 4; ++i) {
    int chunk = tid + i * 128;
    int kk = chunk >> 4;
    int n4 = (chunk & 15) << 2;
#pragma unroll
    for (int j = 0; j < 4; ++j) Bs[(n4 + j) * 32 + kk] = (bf16)t.b[i][j];
  }
}

template <bool RELU, bool RESID>
__global__ __launch_bounds__(128) void gemm128(
    const float* __restrict__ A, int lda, const float* __restrict__ B,
    const float* __restrict__ bias, const float* __restrict__ R,
    float* __restrict__ C, int ldc, int M, int N, int K) {
  __shared__ alignas(16) bf16 As[2][64 * 32];   // [m][k]
  __shared__ alignas(16) bf16 Bs[2][64 * 32];   // [n][k] (transposed)

  const int tid = threadIdx.x;
  const int lane = tid & 31, wave = tid >> 5;
  const int row0 = blockIdx.x * 64;
  const int col0 = blockIdx.y * 64;

  f32x8 acc[4] = {f32x8{}, f32x8{}, f32x8{}, f32x8{}};
  TileRegs regs;

  // prologue: tile 0 -> LDS buf 0
  gemm_load_tile(regs, A, lda, B, N, row0, col0, 0, M, tid);
  gemm_store_tile(regs, As[0], Bs[0], tid);
  __syncthreads();

  const int nk = K >> 5;
  for (int t = 0; t < nk; ++t) {
    const int cur = t & 1;
    if (t + 1 < nk)  // issue next tile's global loads before the math
      gemm_load_tile(regs, A, lda, B, N, row0, col0, (t + 1) << 5, M, tid);

    bf16x16 af = frag_a(As[cur], wave * 16, 32, 0, lane);
#pragma unroll
    for (int nt = 0; nt < 4; ++nt) {
      bf16x16 bf_ = frag_b(Bs[cur], nt * 16, 32, 0, lane);
      acc[nt] = wmma_bf16(af, bf_, acc[nt]);
    }
    if (t + 1 < nk) {
      gemm_store_tile(regs, As[cur ^ 1], Bs[cur ^ 1], tid);
      __syncthreads();
    }
  }

  // Epilogue.  C/D lane layout: row = m0 + (lane>>4)*8 + r, col = lane&15.
  const int mrow = wave * 16 + ((lane >> 4) << 3);
  const int ncol = lane & 15;
#pragma unroll
  for (int nt = 0; nt < 4; ++nt) {
#pragma unroll
    for (int r = 0; r < 8; ++r) {
      int gr = row0 + mrow + r;
      int gc = col0 + nt * 16 + ncol;
      if (gr < M && gc < N) {
        float v = acc[nt][r] + bias[gc];
        if (RESID) v += R[(size_t)gr * ldc + gc];
        if (RELU) v = fmaxf(v, 0.f);
        C[(size_t)gr * ldc + gc] = v;
      }
    }
  }
}

// ---------------------------------------------------------------------------
// Fused relation-aware attention for one (batch, head).  160 threads = 5 waves.
//   S  = q @ k^T                      (WMMA, 5x5 tiles, K=64)
//   qb = q @ ebk_h^T                  (WMMA, 5x1 tiles)
//   score = (S + qb[q, adj]) * scale, pad-mask, softmax
//   w[q,t] = sum_{k: adj==t} attn[q,k]
//   O  = attn @ v + w @ ebv_h         (WMMA, 5x4 tiles, K=96 + K=32)
// Edge-bias head slices (14x64 f32, row stride 1024) are DMA'd into LDS by the
// Tensor Data Mover while the q/k/v vector loads are in flight.
// ---------------------------------------------------------------------------
__global__ __launch_bounds__(160) void attention_kernel(
    const float* __restrict__ Q, const float* __restrict__ Km,
    const float* __restrict__ V, const float* __restrict__ ebk,
    const float* __restrict__ ebv, const int* __restrict__ adj,
    const int* __restrict__ nn, float* __restrict__ O) {
  __shared__ alignas(16) bf16 qs[LPAD * HDIM];    // [q][d]
  __shared__ alignas(16) bf16 ks[LPAD * HDIM];    // [k][d] == B^T for q@k^T
  __shared__ alignas(16) bf16 ebks[16 * HDIM];    // [t][d] == B^T for q@ebk^T
  __shared__ alignas(16) bf16 vT[HDIM * 96];      // [d][k] == B^T for attn@v
  __shared__ alignas(16) bf16 ebvT[HDIM * 32];    // [d][t] == B^T for w@ebv
  __shared__ alignas(16) bf16 att[LPAD * 96];     // [q][k]  (A for attn@v)
  __shared__ alignas(16) bf16 wts[LPAD * 32];     // [q][t]  (A for w@ebv)
  __shared__ float S[LPAD * LPAD];
  __shared__ float qb[LPAD * 16];
  __shared__ alignas(16) float ebk_f[14 * HDIM];  // TDM staging (f32)
  __shared__ alignas(16) float ebv_f[14 * HDIM];  // TDM staging (f32)

  const int b = blockIdx.x, hh = blockIdx.y;
  const int tid = threadIdx.x, lane = tid & 31, wave = tid >> 5;
  const size_t rowbase = (size_t)b * LTOK * EFFD + (size_t)hh * HDIM;

  // Wave 0: kick off the two TDM tile loads (async; tracked by TENSORcnt).
  if (wave == 0) {
    tdm_load_2d_f32((unsigned)(size_t)&ebk_f[0], ebk + (size_t)hh * HDIM,
                    HDIM, 14, EFFD);
    tdm_load_2d_f32((unsigned)(size_t)&ebv_f[0], ebv + (size_t)hh * HDIM,
                    HDIM, 14, EFFD);
  }

  // Meanwhile: vector loads of q/k/v slices, converted to bf16 in LDS.
  for (int i = tid; i < LPAD * HDIM; i += 160) {
    int r = i >> 6, c = i & 63;
    float qv = 0.f, kv = 0.f;
    if (r < LTOK) {
      qv = Q[rowbase + (size_t)r * EFFD + c];
      kv = Km[rowbase + (size_t)r * EFFD + c];
    }
    qs[i] = (bf16)qv; ks[i] = (bf16)kv;
  }
  for (int i = tid; i < HDIM * 96; i += 160) {
    int d = i / 96, kk = i % 96;
    float vv = (kk < LTOK) ? V[rowbase + (size_t)kk * EFFD + d] : 0.f;
    vT[i] = (bf16)vv;
  }

  // Wait for the TDM transfers, then let every wave see the staged data.
  if (wave == 0) __builtin_amdgcn_s_wait_tensorcnt(0);
  __syncthreads();

  // Convert staged f32 bias tiles into the bf16 WMMA operand layouts.
  for (int i = tid; i < 16 * HDIM; i += 160) {
    int t = i >> 6, d = i & 63;
    ebks[i] = (bf16)((t < 14) ? ebk_f[t * HDIM + d] : 0.f);
  }
  for (int i = tid; i < HDIM * 32; i += 160) {
    int d = i >> 5, t = i & 31;
    ebvT[i] = (bf16)((t < 14) ? ebv_f[t * HDIM + d] : 0.f);
  }
  __syncthreads();

  {  // S = q @ k^T  and  qb = q @ ebk^T
    const int m0 = wave * 16;
    const int mr = m0 + ((lane >> 4) << 3);
#pragma unroll
    for (int nt = 0; nt < 5; ++nt) {
      f32x8 acc = {};
#pragma unroll
      for (int k0 = 0; k0 < HDIM; k0 += 32) {
        bf16x16 af = frag_a(qs, m0, HDIM, k0, lane);
        bf16x16 bf_ = frag_b(ks, nt * 16, HDIM, k0, lane);
        acc = wmma_bf16(af, bf_, acc);
      }
      int nc = nt * 16 + (lane & 15);
#pragma unroll
      for (int r = 0; r < 8; ++r) S[(mr + r) * LPAD + nc] = acc[r];
    }
    f32x8 acc = {};
#pragma unroll
    for (int k0 = 0; k0 < HDIM; k0 += 32) {
      bf16x16 af = frag_a(qs, m0, HDIM, k0, lane);
      bf16x16 bf_ = frag_b(ebks, 0, HDIM, k0, lane);
      acc = wmma_bf16(af, bf_, acc);
    }
    int nc = lane & 15;
#pragma unroll
    for (int r = 0; r < 8; ++r) qb[(mr + r) * 16 + nc] = acc[r];
  }
  __syncthreads();

  // Softmax + edge-weight accumulation (one q-row per thread).
  if (tid < LPAD) {
    const int q = tid;
    if (q < LTOK) {
      const int nh = nn[b * 2 + 0], nt2 = nn[b * 2 + 1];
      const int* arow = adj + ((size_t)b * LTOK + q) * LTOK;
      float mx = -3.0e38f;
      for (int k = 0; k < LTOK; ++k) {
        bool valid = (k == 0) || ((k <= MM) ? (k - 1) < nh : (k - MM - 1) < nt2);
        float s = valid ? (S[q * LPAD + k] + qb[q * 16 + arow[k]]) * SCALE_C
                        : -3.0e38f;
        S[q * LPAD + k] = s;
        mx = fmaxf(mx, s);
      }
      float sum = 0.f;
      for (int k = 0; k < LTOK; ++k) {
        float s = S[q * LPAD + k];
        float e = (s > -1.0e38f) ? __expf(s - mx) : 0.f;
        S[q * LPAD + k] = e;
        sum += e;
      }
      const float inv = 1.f / sum;
      float wacc[14];
      for (int t = 0; t < 14; ++t) wacc[t] = 0.f;
      for (int k = 0; k < LTOK; ++k) {
        float a = S[q * LPAD + k] * inv;
        att[q * 96 + k] = (bf16)a;
        wacc[arow[k]] += a;  // masked keys contribute exactly 0
      }
      for (int k = LTOK; k < 96; ++k) att[q * 96 + k] = (bf16)0.f;
      for (int t = 0; t < 32; ++t)
        wts[q * 32 + t] = (bf16)((t < 14) ? wacc[t] : 0.f);
    } else {
      for (int k = 0; k < 96; ++k) att[q * 96 + k] = (bf16)0.f;
      for (int t = 0; t < 32; ++t) wts[q * 32 + t] = (bf16)0.f;
    }
  }
  __syncthreads();

  {  // O = attn @ v + w @ ebv
    const int m0 = wave * 16;
    const int mr = m0 + ((lane >> 4) << 3);
#pragma unroll
    for (int nt = 0; nt < 4; ++nt) {
      f32x8 acc = {};
#pragma unroll
      for (int k0 = 0; k0 < 96; k0 += 32) {
        bf16x16 af = frag_a(att, m0, 96, k0, lane);
        bf16x16 bf_ = frag_b(vT, nt * 16, 96, k0, lane);
        acc = wmma_bf16(af, bf_, acc);
      }
      {
        bf16x16 af = frag_a(wts, m0, 32, 0, lane);
        bf16x16 bf_ = frag_b(ebvT, nt * 16, 32, 0, lane);
        acc = wmma_bf16(af, bf_, acc);
      }
      int nc = nt * 16 + (lane & 15);
#pragma unroll
      for (int r = 0; r < 8; ++r) {
        int q = mr + r;
        if (q < LTOK) O[rowbase + (size_t)q * EFFD + nc] = acc[r];
      }
    }
  }
}

// ---------------------------------------------------------------------------
// Row LayerNorm over EFFD=1024:  Y[row] = (X[row]-mu)/sqrt(var+1e-5)*g + b
// ---------------------------------------------------------------------------
__global__ __launch_bounds__(256) void layernorm_kernel(
    const float* __restrict__ X, const float* __restrict__ g,
    const float* __restrict__ be, float* __restrict__ Y) {
  __shared__ float s1[256], s2[256];
  const int row = blockIdx.x;
  const float* xr = X + (size_t)row * EFFD;
  float v[4], lsum = 0.f, lsq = 0.f;
#pragma unroll
  for (int i = 0; i < 4; ++i) {
    v[i] = xr[threadIdx.x + i * 256];
    lsum += v[i]; lsq += v[i] * v[i];
  }
  s1[threadIdx.x] = lsum; s2[threadIdx.x] = lsq;
  __syncthreads();
  for (int off = 128; off > 0; off >>= 1) {
    if (threadIdx.x < off) {
      s1[threadIdx.x] += s1[threadIdx.x + off];
      s2[threadIdx.x] += s2[threadIdx.x + off];
    }
    __syncthreads();
  }
  const float mean = s1[0] * (1.f / EFFD);
  const float var = s2[0] * (1.f / EFFD) - mean * mean;
  const float rstd = rsqrtf(var + 1e-5f);
#pragma unroll
  for (int i = 0; i < 4; ++i) {
    int c = threadIdx.x + i * 256;
    Y[(size_t)row * EFFD + c] = (v[i] - mean) * rstd * g[c] + be[c];
  }
}

// ---------------------------------------------------------------------------
// Gathers: target concat input [16,768] and neighbor rows of x [16,65,1024]
// ---------------------------------------------------------------------------
__global__ void build_tgt_in(const float* __restrict__ ef,
                             const float* __restrict__ te,
                             const int* __restrict__ tt,
                             float* __restrict__ tin) {
  const int b = blockIdx.x;
  const int i0 = tt[b * 3 + 0], i2 = tt[b * 3 + 2];
  for (int c = threadIdx.x; c < 3 * DD; c += 256) {
    float v;
    if (c < DD)            v = ef[(size_t)i0 * DD + c];
    else if (c < 2 * DD)   v = ef[(size_t)i2 * DD + (c - DD)];
    else                   v = te[c - 2 * DD];  // type 0
    tin[(size_t)b * 3 * DD + c] = v;
  }
}

__global__ void build_neighbors(const float* __restrict__ ef,
                                const float* __restrict__ rf,
                                const float* __restrict__ te,
                                const int* __restrict__ hnb,
                                const int* __restrict__ tnb,
                                float* __restrict__ x) {
  const int b = blockIdx.x;
  const int j = blockIdx.y + 1;  // token 1..64
  const int* nb; int tei;
  if (j <= MM) { nb = hnb + ((size_t)b * MM + (j - 1)) * 3;      tei = 1; }
  else         { nb = tnb + ((size_t)b * MM + (j - MM - 1)) * 3; tei = 2; }
  const int e0 = nb[0], r1 = nb[1], e2 = nb[2];
  for (int c = threadIdx.x; c < EFFD; c += 256) {
    float v;
    if (c < DD)            v = ef[(size_t)e0 * DD + c];
    else if (c < 2 * DD)   v = rf[(size_t)r1 * DD + (c - DD)];
    else if (c < 3 * DD)   v = ef[(size_t)e2 * DD + (c - 2 * DD)];
    else                   v = te[(size_t)tei * DD + (c - 3 * DD)];
    x[((size_t)b * LTOK + j) * EFFD + c] = v;
  }
}

// ---------------------------------------------------------------------------
extern "C" void kernel_launch(void* const* d_in, const int* in_sizes, int n_in,
                              void* d_out, int out_size, void* d_ws,
                              size_t ws_size, hipStream_t stream) {
  const float* ef  = (const float*)d_in[0];
  const float* ebk = (const float*)d_in[1];
  const float* ebv = (const float*)d_in[2];
  auto LP = [&](int l, int j) { return (const float*)d_in[3 + l * 16 + j]; };
  const float* lin_b = (const float*)d_in[67];
  const float* lin_w = (const float*)d_in[68];
  const float* l0_b  = (const float*)d_in[69];
  const float* l0_w  = (const float*)d_in[70];
  const float* rf    = (const float*)d_in[71];
  const float* te    = (const float*)d_in[72];
  const int* tgt  = (const int*)d_in[73];
  const int* hnb  = (const int*)d_in[74];
  const int* tnb  = (const int*)d_in[75];
  const int* nn   = (const int*)d_in[76];
  const int* adj  = (const int*)d_in[77];

  const size_t XSZ = (size_t)BB * LTOK * EFFD;  // 1040*1024
  float* ws  = (float*)d_ws;
  float* x   = ws;            // activations [16,65,1024]
  float* tmp = x + XSZ;       // pre-LN residual sum
  float* qb  = tmp + XSZ;
  float* kb  = qb + XSZ;
  float* vb  = kb + XSZ;
  float* ob  = vb + XSZ;
  float* hb  = ob + XSZ;      // FFN hidden
  float* tin = hb + XSZ;      // [16,768]

  const int MROWS = BB * LTOK;       // 1040
  const int LDX = LTOK * EFFD;       // row stride to hop batch targets in x
  dim3 g17(17, 16);                  // ceil(1040/64) x (1024/64)

  build_tgt_in<<<BB, 256, 0, stream>>>(ef, te, tgt, tin);
  build_neighbors<<<dim3(BB, LTOK - 1), 256, 0, stream>>>(ef, rf, te, hnb, tnb, x);
  // target row of x for each batch: x[b*65] = tgt_in @ linear0_w + b  (ldc=65*1024)
  gemm128<false, false><<<dim3(1, 16), 128, 0, stream>>>(
      tin, 3 * DD, l0_w, l0_b, nullptr, x, LDX, BB, EFFD, 3 * DD);

  for (int l = 0; l < 4; ++l) {
    const float *b1 = LP(l, 0), *b2 = LP(l, 1), *bk_ = LP(l, 2), *bo = LP(l, 3),
                *bq = LP(l, 4), *bv = LP(l, 5), *ln1b = LP(l, 6), *ln1g = LP(l, 7),
                *ln2b = LP(l, 8), *ln2g = LP(l, 9), *w1 = LP(l, 10), *w2 = LP(l, 11),
                *wk = LP(l, 12), *wo = LP(l, 13), *wq = LP(l, 14), *wv = LP(l, 15);

    gemm128<false, false><<<g17, 128, 0, stream>>>(x, EFFD, wq, bq, nullptr, qb,
                                                   EFFD, MROWS, EFFD, EFFD);
    gemm128<false, false><<<g17, 128, 0, stream>>>(x, EFFD, wk, bk_, nullptr, kb,
                                                   EFFD, MROWS, EFFD, EFFD);
    gemm128<false, false><<<g17, 128, 0, stream>>>(x, EFFD, wv, bv, nullptr, vb,
                                                   EFFD, MROWS, EFFD, EFFD);
    attention_kernel<<<dim3(BB, NHEADS), 160, 0, stream>>>(qb, kb, vb, ebk, ebv,
                                                           adj, nn, ob);
    // tmp = x + o@wo + bo ; x = LN1(tmp)
    gemm128<false, true><<<g17, 128, 0, stream>>>(ob, EFFD, wo, bo, x, tmp,
                                                  EFFD, MROWS, EFFD, EFFD);
    layernorm_kernel<<<MROWS, 256, 0, stream>>>(tmp, ln1g, ln1b, x);
    // hb = relu(x@w1 + b1) ; tmp = x + hb@w2 + b2 ; x = LN2(tmp)
    gemm128<true, false><<<g17, 128, 0, stream>>>(x, EFFD, w1, b1, nullptr, hb,
                                                  EFFD, MROWS, EFFD, EFFD);
    gemm128<false, true><<<g17, 128, 0, stream>>>(hb, EFFD, w2, b2, x, tmp,
                                                  EFFD, MROWS, EFFD, EFFD);
    layernorm_kernel<<<MROWS, 256, 0, stream>>>(tmp, ln2g, ln2b, x);
  }

  // out[16,200] = x[:,0,:] @ lin_w + lin_b   (A row stride 65*1024)
  gemm128<false, false><<<dim3(1, 4), 128, 0, stream>>>(
      x, LDX, lin_w, lin_b, nullptr, (float*)d_out, 200, BB, 200, EFFD);
}
